// LocalWindowMultiHeadAttention_11338713661444
// MI455X (gfx1250) — matrix operand order repaired and checked by reading, not compiled, not run on hardware
//
#include <hip/hip_runtime.h>
#include <hip/hip_bf16.h>

// ---------------------------------------------------------------------------
// Local window (7x7, reflect-padded) multi-head attention for MI455X/gfx1250.
//   x[16384,128] -> Q,K,V via bf16 WMMA GEMMs -> per-pixel online-softmax
//   attention (VALU, f32) -> output projection via bf16 WMMA GEMM.
// ---------------------------------------------------------------------------

#define HW    128
#define CDIM  128
#define NPIX  (HW * HW)          // 16384 pixels
#define NHEAD 8
#define HDIM  16                 // head dim
#define RAD   3
#define WIN   7
#define WPOS  (WIN * WIN)        // 49

typedef __attribute__((ext_vector_type(16))) __bf16 v16bf;
typedef __attribute__((ext_vector_type(8)))  float  v8f;

union FragV { uint4 u[2]; v16bf v; };

// ---------------------------------------------------------------------------
// f32 -> bf16 conversion (RNE)
// ---------------------------------------------------------------------------
__global__ __launch_bounds__(256)
void cvt_f32_to_bf16(const float* __restrict__ src,
                     __hip_bfloat16* __restrict__ dst, int n) {
  int i = blockIdx.x * blockDim.x + threadIdx.x;
  if (i < n) dst[i] = __float2bfloat16(src[i]);
}

// ---------------------------------------------------------------------------
// WMMA fragment loaders (layouts per CDNA5 ISA 7.12.2, wave32):
//  A (16x32 bf16): lane holds row M=lane%16; elements 0..7 = K(lane/16)*8..+7,
//                  elements 8..15 = 16+(lane/16)*8..+7  -> two 16B loads.
//  B (32x16 bf16): lane holds col N=lane%16; same K order as A:
//                  elements 0..7 = K(lane/16)*8..+7, 8..15 = 16+(lane/16)*8..+7.
//  C/D (16x16 f32): acc[r] is row M=(lane/16)*8+r, col N=lane%16.
// ---------------------------------------------------------------------------
__device__ inline v16bf load_frag_A(const __hip_bfloat16* __restrict__ rowM,
                                    int kchunk, int lane) {
  const __hip_bfloat16* p = rowM + kchunk * 32 + ((lane >> 4) << 3);
  FragV f;
  f.u[0] = *(const uint4*)(p);
  f.u[1] = *(const uint4*)(p + 16);
  return f.v;
}

__device__ inline v16bf load_frag_B(const __hip_bfloat16* __restrict__ rowN,
                                    int kchunk, int lane) {
  const __hip_bfloat16* p = rowN + kchunk * 32 + ((lane >> 4) << 3);
  FragV f;
  f.u[0] = *(const uint4*)(p);
  f.u[1] = *(const uint4*)(p + 16);
  return f.v;
}

// ---------------------------------------------------------------------------
// Fused Q/K/V projection: out = x @ W.T + b  (M=16384, K=128, N=128) x3.
// One wave per 16-row M tile; A fragments (4 K-chunks) reused across the
// 3 weight matrices x 8 N tiles = 24 reuses. 96 v_wmma per wave.
// ---------------------------------------------------------------------------
__global__ __launch_bounds__(32)
void gemm_qkv(const __hip_bfloat16* __restrict__ xb,
              const __hip_bfloat16* __restrict__ wq, const float* __restrict__ bq,
              const __hip_bfloat16* __restrict__ wk, const float* __restrict__ bk,
              const __hip_bfloat16* __restrict__ wv, const float* __restrict__ bv,
              __hip_bfloat16* __restrict__ qo,
              __hip_bfloat16* __restrict__ ko,
              __hip_bfloat16* __restrict__ vo) {
  const int lane = threadIdx.x;
  const int m0   = blockIdx.x * 16;
  const int mrow = m0 + (lane & 15);
  const int ncol = lane & 15;
  const int mhi  = (lane >> 4) << 3;   // 0 or 8 (row offset in C/D frag)

  v16bf afrag[4];
  const __hip_bfloat16* arow = xb + (size_t)mrow * CDIM;
#pragma unroll
  for (int kc = 0; kc < 4; ++kc) afrag[kc] = load_frag_A(arow, kc, lane);

  const __hip_bfloat16* Ws[3] = {wq, wk, wv};
  const float*          Bs[3] = {bq, bk, bv};
  __hip_bfloat16*       Os[3] = {qo, ko, vo};

#pragma unroll
  for (int wsel = 0; wsel < 3; ++wsel) {
    const __hip_bfloat16* W = Ws[wsel];
    const float*          B = Bs[wsel];
    __hip_bfloat16*       O = Os[wsel];
#pragma unroll
    for (int nt = 0; nt < 8; ++nt) {
      const int n = nt * 16 + ncol;
      const __hip_bfloat16* wrow = W + (size_t)n * CDIM;  // row n of W == B col n
      v8f acc = {};
#pragma unroll
      for (int kc = 0; kc < 4; ++kc) {
        v16bf bfrag = load_frag_B(wrow, kc, lane);
        acc = __builtin_amdgcn_wmma_f32_16x16x32_bf16(
            false, afrag[kc], false, bfrag, (short)0, acc, false, false);
      }
      const float bias = B[n];
#pragma unroll
      for (int r = 0; r < 8; ++r) {
        const int mr = m0 + mhi + r;
        O[(size_t)mr * CDIM + n] = __float2bfloat16(acc[r] + bias);
      }
    }
  }
}

// ---------------------------------------------------------------------------
// Output projection: out_f32 = ctx @ Wo.T + bo  (fully overwrites d_out).
// ---------------------------------------------------------------------------
__global__ __launch_bounds__(32)
void gemm_out(const __hip_bfloat16* __restrict__ ctxb,
              const __hip_bfloat16* __restrict__ wo, const float* __restrict__ bo,
              float* __restrict__ out) {
  const int lane = threadIdx.x;
  const int m0   = blockIdx.x * 16;
  const int mrow = m0 + (lane & 15);
  const int ncol = lane & 15;
  const int mhi  = (lane >> 4) << 3;

  v16bf afrag[4];
  const __hip_bfloat16* arow = ctxb + (size_t)mrow * CDIM;
#pragma unroll
  for (int kc = 0; kc < 4; ++kc) afrag[kc] = load_frag_A(arow, kc, lane);

#pragma unroll
  for (int nt = 0; nt < 8; ++nt) {
    const int n = nt * 16 + ncol;
    const __hip_bfloat16* wrow = wo + (size_t)n * CDIM;
    v8f acc = {};
#pragma unroll
    for (int kc = 0; kc < 4; ++kc) {
      v16bf bfrag = load_frag_B(wrow, kc, lane);
      acc = __builtin_amdgcn_wmma_f32_16x16x32_bf16(
          false, afrag[kc], false, bfrag, (short)0, acc, false, false);
    }
    const float bias = bo[n];
#pragma unroll
    for (int r = 0; r < 8; ++r) {
      const int mr = m0 + mhi + r;
      out[(size_t)mr * CDIM + n] = acc[r] + bias;
    }
  }
}

// ---------------------------------------------------------------------------
// Per-pixel 7x7 local attention with reflect indexing + online softmax.
// Block = 128 threads (1 thread per channel, 4 wave32s). Each head's 16
// channels sit inside one 16-lane half-wave -> shfl_xor(1,2,4,8) reduction,
// softmax state replicated per half-wave. No LDS, no barriers.
// ---------------------------------------------------------------------------
__global__ __launch_bounds__(128)
void attn_win7(const __hip_bfloat16* __restrict__ qb,
               const __hip_bfloat16* __restrict__ kb,
               const __hip_bfloat16* __restrict__ vb,
               __hip_bfloat16* __restrict__ ctxb) {
  const int pix = blockIdx.x;
  const int y = pix >> 7, x = pix & (HW - 1);
  const int c = threadIdx.x;

  // fold scale = 1/sqrt(16) = 0.25 into q
  const float qc = __bfloat162float(qb[(size_t)pix * CDIM + c]) * 0.25f;

  float m = -1e30f, l = 0.0f, acc = 0.0f;

#pragma unroll 1
  for (int w = 0; w < WPOS; ++w) {
    const int dy = w / WIN - RAD;
    const int dx = w % WIN - RAD;
    int yy = y + dy; yy = (yy < 0) ? -yy : yy; yy = (yy > HW - 1) ? 2 * (HW - 1) - yy : yy;
    int xx = x + dx; xx = (xx < 0) ? -xx : xx; xx = (xx > HW - 1) ? 2 * (HW - 1) - xx : xx;
    const size_t idx = (size_t)((yy << 7) + xx) * CDIM + c;

    float s = qc * __bfloat162float(kb[idx]);
    s += __shfl_xor(s, 1, 32);
    s += __shfl_xor(s, 2, 32);
    s += __shfl_xor(s, 4, 32);
    s += __shfl_xor(s, 8, 32);          // full head-dim dot, replicated in 16 lanes

    const float mnew = fmaxf(m, s);
    const float corr = __expf(m - mnew);
    const float p    = __expf(s - mnew);
    l   = l * corr + p;
    acc = acc * corr + p * __bfloat162float(vb[idx]);
    m   = mnew;
  }

  ctxb[(size_t)pix * CDIM + c] = __float2bfloat16(acc / l);
}

// ---------------------------------------------------------------------------
// Host launcher
// ---------------------------------------------------------------------------
extern "C" void kernel_launch(void* const* d_in, const int* in_sizes, int n_in,
                              void* d_out, int out_size, void* d_ws, size_t ws_size,
                              hipStream_t stream) {
  (void)in_sizes; (void)n_in; (void)out_size; (void)ws_size;

  const float* x  = (const float*)d_in[0];
  const float* Wq = (const float*)d_in[1];
  const float* bq = (const float*)d_in[2];
  const float* Wk = (const float*)d_in[3];
  const float* bk = (const float*)d_in[4];
  const float* Wv = (const float*)d_in[5];
  const float* bv = (const float*)d_in[6];
  const float* Wo = (const float*)d_in[7];
  const float* bo = (const float*)d_in[8];
  float* out = (float*)d_out;

  char* ws = (char*)d_ws;
  const size_t SZ = (size_t)NPIX * CDIM * sizeof(__hip_bfloat16);  // 4 MiB
  __hip_bfloat16* xb   = (__hip_bfloat16*)(ws + 0 * SZ);
  __hip_bfloat16* qbuf = (__hip_bfloat16*)(ws + 1 * SZ);
  __hip_bfloat16* kbuf = (__hip_bfloat16*)(ws + 2 * SZ);
  __hip_bfloat16* vbuf = (__hip_bfloat16*)(ws + 3 * SZ);
  __hip_bfloat16* cbuf = (__hip_bfloat16*)(ws + 4 * SZ);
  __hip_bfloat16* wqb  = (__hip_bfloat16*)(ws + 5 * SZ);
  __hip_bfloat16* wkb  = wqb + CDIM * CDIM;
  __hip_bfloat16* wvb  = wkb + CDIM * CDIM;
  __hip_bfloat16* wob  = wvb + CDIM * CDIM;

  // 1) convert activations + weights to bf16
  cvt_f32_to_bf16<<<(NPIX * CDIM + 255) / 256, 256, 0, stream>>>(x, xb, NPIX * CDIM);
  cvt_f32_to_bf16<<<(CDIM * CDIM + 255) / 256, 256, 0, stream>>>(Wq, wqb, CDIM * CDIM);
  cvt_f32_to_bf16<<<(CDIM * CDIM + 255) / 256, 256, 0, stream>>>(Wk, wkb, CDIM * CDIM);
  cvt_f32_to_bf16<<<(CDIM * CDIM + 255) / 256, 256, 0, stream>>>(Wv, wvb, CDIM * CDIM);
  cvt_f32_to_bf16<<<(CDIM * CDIM + 255) / 256, 256, 0, stream>>>(Wo, wob, CDIM * CDIM);

  // 2) fused QKV projection (WMMA)
  gemm_qkv<<<NPIX / 16, 32, 0, stream>>>(xb, wqb, bq, wkb, bk, wvb, bv,
                                         qbuf, kbuf, vbuf);

  // 3) local-window attention, online softmax
  attn_win7<<<NPIX, 128, 0, stream>>>(qbuf, kbuf, vbuf, cbuf);

  // 4) output projection (WMMA) -> f32 d_out
  gemm_out<<<NPIX / 16, 32, 0, stream>>>(cbuf, wob, bo, out);
}
